// InBatchSampledSoftmaxLoss_63041529971024
// MI455X (gfx1250) — matrix operand order repaired and checked by reading, not compiled, not run on hardware
//
#include <hip/hip_runtime.h>
#include <hip/hip_bf16.h>
#include <math.h>

// CDNA5 / gfx1250: wave32, WMMA 16x16x32 f16->f32.
typedef _Float16 v16h __attribute__((ext_vector_type(16)));
typedef _Float16 v8h  __attribute__((ext_vector_type(8)));
typedef float    v8f  __attribute__((ext_vector_type(8)));

#define T_ROWS   8192          // B*(N-1) = 8*1024
#define DDIM     64
#define TEMP_INV 20.0f         // 1 / 0.05
#define LMAX     20.5f         // fixed softmax max: |logit| <= 20 (unit vectors) + slack
#define LDS_STRIDE 72          // 64 halves + 8 pad (144B row) to stagger LDS banks
#define COL_SPLIT 4            // blocks along the column (j) dimension
#define COLS_PER_BLK (T_ROWS / COL_SPLIT)

// ---- workspace layout (offsets in bytes) ----
// accum[2]  @ 0        : [0]=sum of valid diag logp, [1]=valid count
// vscale    @ 256      : 8192 f32 column-valid scales (1.0/0.0)
// diagv     @ 256+32K  : 8192 f32 diagonal logits (fp32, matches reference)
// rowsum    @ 256+64K  : 8192 f32 per-row sum of exp(s - LMAX) over valid cols
// qh        @ 256+96K  : 8192x64 f16 normalized q
// kh        : qh + 1M: 8192x64 f16 normalized k

__global__ void init_ws_kernel(float* accum, float* rowsum) {
    const int i = blockIdx.x * blockDim.x + threadIdx.x;
    if (i < T_ROWS) rowsum[i] = 0.0f;
    if (i < 2) accum[i] = 0.0f;
}

// Normalize q = out[:, :-1], k = tgt[:, 1:], convert to f16, emit column valid
// scale, fp32 diagonal logit, and accumulate the valid count.
__global__ __launch_bounds__(256) void prep_kernel(
        const float* __restrict__ outp, const float* __restrict__ targ,
        const unsigned char* __restrict__ mask, float* __restrict__ accum,
        float* __restrict__ vscale, float* __restrict__ diagv,
        _Float16* __restrict__ qh, _Float16* __restrict__ kh) {
    const int i = blockIdx.x * blockDim.x + threadIdx.x;   // 0..8191
    const int b = i >> 10, n = i & 1023;
    const float* qrow = outp + (size_t)(b * 1025 + n) * DDIM;
    const float* krow = targ + (size_t)(b * 1025 + n + 1) * DDIM;

    float qss = 0.f, kss = 0.f, qkd = 0.f;
#pragma unroll
    for (int d = 0; d < DDIM; d += 4) {
        float4 a = *(const float4*)(qrow + d);
        float4 c = *(const float4*)(krow + d);
        qss += a.x*a.x + a.y*a.y + a.z*a.z + a.w*a.w;
        kss += c.x*c.x + c.y*c.y + c.z*c.z + c.w*c.w;
        qkd += a.x*c.x + a.y*c.y + a.z*c.z + a.w*c.w;
    }
    const float qinv = 1.f / fmaxf(sqrtf(qss), 1e-12f);
    const float kinv = 1.f / fmaxf(sqrtf(kss), 1e-12f);
    diagv[i] = qkd * qinv * kinv * TEMP_INV;   // fp32 diagonal logit

#pragma unroll
    for (int d = 0; d < DDIM; d += 8) {
        float4 a0 = *(const float4*)(qrow + d);
        float4 a1 = *(const float4*)(qrow + d + 4);
        float4 c0 = *(const float4*)(krow + d);
        float4 c1 = *(const float4*)(krow + d + 4);
        float qt[8] = {a0.x,a0.y,a0.z,a0.w,a1.x,a1.y,a1.z,a1.w};
        float kt[8] = {c0.x,c0.y,c0.z,c0.w,c1.x,c1.y,c1.z,c1.w};
        v8h hq, hk;
#pragma unroll
        for (int e = 0; e < 8; e++) {
            hq[e] = (_Float16)(qt[e] * qinv);
            hk[e] = (_Float16)(kt[e] * kinv);
        }
        *(v8h*)(qh + (size_t)i * DDIM + d) = hq;
        *(v8h*)(kh + (size_t)i * DDIM + d) = hk;
    }

    const float v = mask[b * 1025 + n + 1] ? 1.0f : 0.0f;  // valid = mask[:,1:]
    vscale[i] = v;
    float tot = v;
#pragma unroll
    for (int off = 16; off > 0; off >>= 1) tot += __shfl_xor(tot, off, 32);
    if ((threadIdx.x & 31) == 0) atomicAdd(accum + 1, tot);
}

// Streaming pass: grid (64 row-blocks x COL_SPLIT col-blocks). Each wave owns
// 16 rows; the block streams 128-col k-tiles through LDS; each 16x16 score
// tile = 2x V_WMMA_F32_16X16X32_F16 (K=64). Fixed-max exp accumulation; each
// block commits one atomicAdd of its partial exp-sum per row.
__global__ __launch_bounds__(256) void sumexp_kernel(
        const _Float16* __restrict__ qh, const _Float16* __restrict__ kh,
        const float* __restrict__ vscale, float* __restrict__ rowsum) {
    __shared__ _Float16 lds[128 * LDS_STRIDE];   // 18 KB k-tile

    const int lane   = threadIdx.x & 31;
    const int wave   = threadIdx.x >> 5;
    const int rblock = blockIdx.x * 128 + wave * 16;
    const int m16    = lane & 15;
    const int jbase  = blockIdx.y * COLS_PER_BLK;

    // A fragments: contiguous k order (k = 16h + s), same as B.
    // lane<16 -> M=lane, K in kb..kb+15, kb=0; lane>=16 -> kb=16.
    const int kbase = (lane < 16) ? 0 : 16;
    const _Float16* qrow = qh + (size_t)(rblock + m16) * DDIM;
    v16h a0, a1;
    {
        v8h lo0 = *(const v8h*)(qrow + kbase);
        v8h hi0 = *(const v8h*)(qrow + kbase + 8);
        v8h lo1 = *(const v8h*)(qrow + kbase + 32);
        v8h hi1 = *(const v8h*)(qrow + kbase + 40);
#pragma unroll
        for (int e = 0; e < 8; e++) {
            a0[e] = lo0[e]; a0[8 + e] = hi0[e];
            a1[e] = lo1[e]; a1[8 + e] = hi1[e];
        }
    }
    // B fragments: same contiguous k order as A: lane<16 -> col=lane, K=0..15;
    // lane>=16 -> col=lane-16, K=16..31 (one contiguous 16-half run).
    const int kboff = (lane < 16) ? 0 : 16;

    float psum[8];
#pragma unroll
    for (int r = 0; r < 8; r++) psum[r] = 0.f;

    const int cS = threadIdx.x >> 1;                 // staging: 2 threads / col
    const int pS = (threadIdx.x & 1) * 32;

    for (int j0 = jbase; j0 < jbase + COLS_PER_BLK; j0 += 128) {
        {   // stage 128x64 f16 k-tile -> LDS (L2-resident source)
            const _Float16* src = kh + (size_t)(j0 + cS) * DDIM + pS;
            _Float16* dst = lds + cS * LDS_STRIDE + pS;
#pragma unroll
            for (int e = 0; e < 32; e += 8)
                *(v8h*)(dst + e) = *(const v8h*)(src + e);
        }
        if (j0 + 128 < jbase + COLS_PER_BLK)   // hint next tile -> global_prefetch_b8
            __builtin_prefetch(kh + (size_t)(j0 + 128 + cS) * DDIM + pS, 0, 1);
        __syncthreads();

#pragma unroll
        for (int st = 0; st < 8; st++) {
            const _Float16* kc = lds + (st * 16 + m16) * LDS_STRIDE + kboff;
            v16h b0 = *(const v16h*)(kc);        // K 0..31 slice
            v16h b1 = *(const v16h*)(kc + 32);   // K 32..63 slice
            v8f s = {};
            s = __builtin_amdgcn_wmma_f32_16x16x32_f16(false, a0, false, b0,
                                                       (short)0, s, false, false);
            s = __builtin_amdgcn_wmma_f32_16x16x32_f16(false, a1, false, b1,
                                                       (short)0, s, false, false);
            // C/D layout: vgpr r, lane<16: (M=r, N=lane); lane>=16: (M=8+r, N=lane-16)
            const float cs = vscale[j0 + st * 16 + m16];   // column valid scale
#pragma unroll
            for (int r = 0; r < 8; r++)
                psum[r] += cs * __expf(s[r] * TEMP_INV - LMAX);
        }
        __syncthreads();
    }

    // Reduce partial exp-sums across each 16-lane half (same M set per half).
#pragma unroll
    for (int off = 1; off < 16; off <<= 1) {
#pragma unroll
        for (int r = 0; r < 8; r++)
            psum[r] += __shfl_xor(psum[r], off, 32);
    }
    // Lanes 0..7 own rows M=0..7; lanes 24..31 own rows M=8..15.
    const int m = (lane < 8) ? lane : ((lane >= 24) ? (lane - 16) : -1);
    if (m >= 0) atomicAdd(rowsum + rblock + m, psum[m & 7]);
}

// Per-row: logp_ii = diag - LMAX - log(rowsum); average over valid rows.
__global__ __launch_bounds__(256) void finalize_kernel(
        const float* __restrict__ vscale, const float* __restrict__ diagv,
        const float* __restrict__ rowsum, float* __restrict__ accum) {
    const int i = blockIdx.x * blockDim.x + threadIdx.x;   // 0..8191
    float contrib = 0.f;
    const float rowv = vscale[i];
    if (rowv != 0.f) contrib = diagv[i] - LMAX - __logf(rowsum[i]);
#pragma unroll
    for (int off = 16; off > 0; off >>= 1) contrib += __shfl_xor(contrib, off, 32);
    if ((threadIdx.x & 31) == 0) atomicAdd(accum, contrib);
}

__global__ void finish_kernel(const float* __restrict__ accum,
                              float* __restrict__ out) {
    if (threadIdx.x == 0 && blockIdx.x == 0)
        out[0] = -accum[0] / accum[1];
}

extern "C" void kernel_launch(void* const* d_in, const int* in_sizes, int n_in,
                              void* d_out, int out_size, void* d_ws, size_t ws_size,
                              hipStream_t stream) {
    (void)in_sizes; (void)n_in; (void)out_size; (void)ws_size;
    const float*         outp = (const float*)d_in[0];          // (8,1025,64) f32
    const float*         targ = (const float*)d_in[1];          // (8,1025,64) f32
    const unsigned char* mask = (const unsigned char*)d_in[2];  // (8,1025) bool

    char* ws = (char*)d_ws;
    float*     accum  = (float*)ws;
    float*     vscale = (float*)(ws + 256);
    float*     diagv  = (float*)(ws + 256 + 32768);
    float*     rowsum = (float*)(ws + 256 + 65536);
    _Float16*  qh     = (_Float16*)(ws + 256 + 98304);
    _Float16*  kh     = qh + (size_t)T_ROWS * DDIM;

    init_ws_kernel<<<T_ROWS / 256, 256, 0, stream>>>(accum, rowsum);
    prep_kernel<<<T_ROWS / 256, 256, 0, stream>>>(outp, targ, mask, accum,
                                                  vscale, diagv, qh, kh);
    sumexp_kernel<<<dim3(T_ROWS / 128, COL_SPLIT), 256, 0, stream>>>(qh, kh,
                                                  vscale, rowsum);
    finalize_kernel<<<T_ROWS / 256, 256, 0, stream>>>(vscale, diagv, rowsum,
                                                      accum);
    finish_kernel<<<1, 32, 0, stream>>>(accum, (float*)d_out);
}
